// GCN_28346784154177
// MI455X (gfx1250) — matrix operand order repaired and checked by reading, not compiled, not run on hardware
//
#include <hip/hip_runtime.h>
#include <hip/hip_bf16.h>

// ---------------------------------------------------------------------------
// GCN forward for MI455X (gfx1250, wave32, WMMA).
//   h1 = relu(spmm(x @ W1) + b1); out = log_softmax(spmm(h1 @ W2) + b2)
// GEMMs: V_WMMA_F32_16X16X32_BF16 (bf16 in, f32 accum), A-tiles staged in LDS
// (double-buffered) so each x row is fetched from HBM once per workgroup.
// SpMM: global_atomic_add_f32 (XW1/AGG1 are L2-resident: 102 MB vs 192 MB L2),
// scalarized edge metadata / non-temporal edge streams.
// ---------------------------------------------------------------------------

#define M_NODES 100000
#define N_EDGES 3200000
#define K1 512      // NFEAT
#define N1 256      // NHID
#define K2 256
#define N2 40       // NCLASS
#define N2P 48      // padded to 3 WMMA n-tiles

#define LDS_STRIDE 18   // u32 pairs per A-tile row (16 + pad, conflict-free reads)

typedef __attribute__((ext_vector_type(16))) __bf16 v16bf;
typedef __attribute__((ext_vector_type(8)))  float  v8f;

union Frag16 {               // 8 VGPRs = 16 bf16 elements (A or B operand)
    unsigned int u[8];
    v16bf v;
};

// round-to-nearest-even f32 -> bf16, packed pair {lo=a, hi=b}
__device__ __forceinline__ unsigned int pk_bf16(float a, float b) {
    union { float f; unsigned int u; } ua, ub;
    ua.f = a; ub.f = b;
    unsigned int r0 = (ua.u + 0x7FFFu + ((ua.u >> 16) & 1u)) >> 16;
    unsigned int r1 = (ub.u + 0x7FFFu + ((ub.u >> 16) & 1u)) >> 16;
    return (r1 << 16) | (r0 & 0xFFFFu);
}

__device__ __forceinline__ unsigned short bf16_1(float a) {
    union { float f; unsigned int u; } ua; ua.f = a;
    return (unsigned short)((ua.u + 0x7FFFu + ((ua.u >> 16) & 1u)) >> 16);
}

// A-operand (16x32 bf16) K-pair index for VGPR v, lane-half g (0/1):
// lanes 0-15 hold K 0..15 (contiguous); lanes 16-31 hold K 16..31.
__device__ __forceinline__ int a_pair_idx(int v, int g) {
    return v + 8 * g;
}

// --------------------------- weight conversion -----------------------------
// W1t[n][k] = bf16(W1[k][n]) : [256][512]
__global__ __launch_bounds__(256) void conv_w1_kernel(
    const float* __restrict__ w1, unsigned short* __restrict__ w1t) {
    int gid = blockIdx.x * 256 + threadIdx.x;            // 131072 threads
    int n = gid >> 9, k = gid & 511;
    w1t[gid] = bf16_1(w1[(size_t)k * N1 + n]);
}

// W2t[n][k] = bf16(W2[k][n]) padded to 48 rows : [48][256]
__global__ __launch_bounds__(256) void conv_w2_kernel(
    const float* __restrict__ w2, unsigned short* __restrict__ w2t) {
    int gid = blockIdx.x * 256 + threadIdx.x;            // 12288 threads
    int n = gid >> 8, k = gid & 255;
    float v = (n < N2) ? w2[(size_t)k * N2 + n] : 0.0f;
    w2t[gid] = bf16_1(v);
}

// ------------------------------- GEMM 1 ------------------------------------
// XW1[M][256] = x[M][512] @ W1. WG = 8 waves = 2 m-strips x 4 n-strips,
// WG tile 64m x 256n. A-tile (64m x 32k) converted to bf16 once and shared
// via LDS (double-buffered, one barrier per K-step).
__global__ __launch_bounds__(256) void gemm1_kernel(
    const float* __restrict__ x,
    const unsigned int* __restrict__ w1t,     // [256][256] u32 K-pairs
    float* __restrict__ xw1) {
    __shared__ unsigned int atile[2][64 * LDS_STRIDE];   // 9216 B

    const int tid  = threadIdx.x;
    const int lane = tid & 31, wave = tid >> 5;
    const int g = lane >> 4, l16 = lane & 15;
    const int ms = wave >> 2;                 // m-strip in WG (0/1)
    const int n0 = (wave & 3) * 64;           // n-strip base
    const int mblk = blockIdx.x * 64;

    // ---- loader role: thread t -> row (t>>2), quad (t&3); 32 k per step ----
    const int lm = tid >> 2, lq = tid & 3;
    int mg = mblk + lm;
    if (mg >= M_NODES) mg = M_NODES - 1;      // clamp (stores are guarded)
    const float* xrow = x + (size_t)mg * K1;

    // prologue: stage k-step 0 into buffer 0
    {
        float4 c0 = *(const float4*)(xrow + 4 * lq);
        float4 c1 = *(const float4*)(xrow + 16 + 4 * lq);
        unsigned int* ar = &atile[0][lm * LDS_STRIDE];
        ar[2 * lq]         = pk_bf16(c0.x, c0.y);
        ar[2 * lq + 1]     = pk_bf16(c0.z, c0.w);
        ar[8 + 2 * lq]     = pk_bf16(c1.x, c1.y);
        ar[8 + 2 * lq + 1] = pk_bf16(c1.z, c1.w);
    }

    v8f acc[2][4] = {};
    int buf = 0;
    for (int kk = 0; kk < 16; ++kk) {         // K1/32 steps
        __syncthreads();
        // prefetch next A-tile into registers while we compute
        float4 p0, p1;
        if (kk < 15) {
            p0 = *(const float4*)(xrow + (kk + 1) * 32 + 4 * lq);
            p1 = *(const float4*)(xrow + (kk + 1) * 32 + 16 + 4 * lq);
        }

        Frag16 a[2];
#pragma unroll
        for (int mt = 0; mt < 2; ++mt) {
            const unsigned int* arow =
                &atile[buf][(ms * 32 + mt * 16 + l16) * LDS_STRIDE];
#pragma unroll
            for (int v = 0; v < 8; ++v) a[mt].u[v] = arow[a_pair_idx(v, g)];
        }
#pragma unroll
        for (int nt = 0; nt < 4; ++nt) {
            Frag16 b;
            const unsigned int* wr =
                w1t + (size_t)(n0 + nt * 16 + l16) * (K1 / 2) + kk * 16 + 8 * g;
#pragma unroll
            for (int v = 0; v < 8; ++v) b.u[v] = wr[v];
#pragma unroll
            for (int mt = 0; mt < 2; ++mt)
                acc[mt][nt] = __builtin_amdgcn_wmma_f32_16x16x32_bf16(
                    false, a[mt].v, false, b.v, (short)0, acc[mt][nt], false, false);
        }

        if (kk < 15) {
            unsigned int* ar = &atile[buf ^ 1][lm * LDS_STRIDE];
            ar[2 * lq]         = pk_bf16(p0.x, p0.y);
            ar[2 * lq + 1]     = pk_bf16(p0.z, p0.w);
            ar[8 + 2 * lq]     = pk_bf16(p1.x, p1.y);
            ar[8 + 2 * lq + 1] = pk_bf16(p1.z, p1.w);
        }
        buf ^= 1;
    }

    // C layout: VGPR r -> M = r + 8*g ; lane%16 -> N
    const int m0 = mblk + ms * 32;
#pragma unroll
    for (int mt = 0; mt < 2; ++mt)
#pragma unroll
        for (int nt = 0; nt < 4; ++nt)
#pragma unroll
            for (int r = 0; r < 8; ++r) {
                int m = m0 + mt * 16 + r + 8 * g;
                int n = n0 + nt * 16 + l16;
                if (m < M_NODES) xw1[(size_t)m * N1 + n] = acc[mt][nt][r];
            }
}

// ------------------------------ SpMM (256ch) --------------------------------
// agg1[dst[e]] += w[e] * h[src[e]] ; 64 threads per edge (wave-uniform edge:
// metadata loads scalarize through the constant cache), float4 per thread.
__global__ __launch_bounds__(256) void spmm256_kernel(
    const int* __restrict__ src, const int* __restrict__ dst,
    const float* __restrict__ ew, const float* __restrict__ h,
    float* __restrict__ out) {
    // 4 edges per 256-thread block; 800000 blocks covers all 3.2M edges exactly
    const int e = __builtin_amdgcn_readfirstlane(blockIdx.x * 4 + (threadIdx.x >> 6));
    const float w = ew[e];
    const int s = src[e], d = dst[e];
    const int c = (threadIdx.x & 63) * 4;
    const float4 hv = *(const float4*)(h + (size_t)s * N1 + c);
    float* o = out + (size_t)d * N1 + c;
    atomicAdd(o + 0, w * hv.x);
    atomicAdd(o + 1, w * hv.y);
    atomicAdd(o + 2, w * hv.z);
    atomicAdd(o + 3, w * hv.w);
}

// --------------------- bias + relu + pack to bf16 pairs ---------------------
__global__ __launch_bounds__(256) void bias_relu_pack_kernel(
    const float* __restrict__ agg1, const float* __restrict__ b1,
    unsigned int* __restrict__ h1) {            // h1: [M][128] u32 K-pairs
    long long gid = (long long)blockIdx.x * 256 + threadIdx.x;
    if (gid >= (long long)M_NODES * 128) return;
    int m = (int)(gid >> 7), p = (int)gid & 127, kk = p * 2;
    const float2 v = *(const float2*)(agg1 + (size_t)m * N1 + kk);
    float a = fmaxf(v.x + b1[kk], 0.0f);
    float b = fmaxf(v.y + b1[kk + 1], 0.0f);
    h1[(size_t)m * 128 + p] = pk_bf16(a, b);
}

// ------------------------------- GEMM 2 ------------------------------------
// XW2[M][40] = h1[M][256] @ W2. Wave tile: 32m x 48n (2x3 tiles), 8 waves/WG.
// A operand is already bf16-packed: direct u32 pair loads, no conversion.
__global__ __launch_bounds__(256) void gemm2_kernel(
    const unsigned int* __restrict__ h1,      // [M][128] u32 K-pairs (bf16)
    const unsigned int* __restrict__ w2t,     // [48][128] u32 K-pairs
    float* __restrict__ xw2) {
    const int tid  = threadIdx.x;
    const int lane = tid & 31, wave = tid >> 5;
    const int g = lane >> 4, l16 = lane & 15;
    const int m0 = (blockIdx.x * 8 + wave) * 32;
    if (m0 >= M_NODES) return;                // whole-wave uniform, no barriers here

    v8f acc[2][3] = {};
    for (int k0 = 0; k0 < K2; k0 += 32) {
        Frag16 a[2];
#pragma unroll
        for (int mt = 0; mt < 2; ++mt) {
            const unsigned int* hr =
                h1 + (size_t)(m0 + mt * 16 + l16) * (K2 / 2) + (k0 >> 1);
#pragma unroll
            for (int v = 0; v < 8; ++v) a[mt].u[v] = hr[a_pair_idx(v, g)];
        }
#pragma unroll
        for (int nt = 0; nt < 3; ++nt) {
            Frag16 b;
            const unsigned int* wr =
                w2t + (size_t)(nt * 16 + l16) * (K2 / 2) + (k0 >> 1) + 8 * g;
#pragma unroll
            for (int v = 0; v < 8; ++v) b.u[v] = wr[v];
#pragma unroll
            for (int mt = 0; mt < 2; ++mt)
                acc[mt][nt] = __builtin_amdgcn_wmma_f32_16x16x32_bf16(
                    false, a[mt].v, false, b.v, (short)0, acc[mt][nt], false, false);
        }
    }
#pragma unroll
    for (int mt = 0; mt < 2; ++mt)
#pragma unroll
        for (int nt = 0; nt < 3; ++nt)
#pragma unroll
            for (int r = 0; r < 8; ++r) {
                int m = m0 + mt * 16 + r + 8 * g;
                int n = nt * 16 + l16;
                if (n < N2) xw2[(size_t)m * N2 + n] = acc[mt][nt][r];
            }
}

// ------------------------------ SpMM (40ch) ---------------------------------
// 10 threads/edge (not wave-uniform): keep vector loads but stream the edge
// arrays non-temporally so they don't evict L2-resident node data.
__global__ __launch_bounds__(256) void spmm40_kernel(
    const int* __restrict__ src, const int* __restrict__ dst,
    const float* __restrict__ ew, const float* __restrict__ h,
    float* __restrict__ out) {
    long long gid = (long long)blockIdx.x * 256 + threadIdx.x;
    if (gid >= (long long)N_EDGES * 10) return;
    int e = (int)(gid / 10);
    int c = (int)(gid % 10) * 4;
    const float w = __builtin_nontemporal_load(ew + e);
    const int s = __builtin_nontemporal_load(src + e);
    const int d = __builtin_nontemporal_load(dst + e);
    const float4 hv = *(const float4*)(h + (size_t)s * N2 + c);
    float* o = out + (size_t)d * N2 + c;
    atomicAdd(o + 0, w * hv.x);
    atomicAdd(o + 1, w * hv.y);
    atomicAdd(o + 2, w * hv.z);
    atomicAdd(o + 3, w * hv.w);
}

// ------------------------ bias + log_softmax (in place) ---------------------
__global__ __launch_bounds__(256) void logsoftmax_kernel(
    float* __restrict__ out, const float* __restrict__ b2) {
    int m = blockIdx.x * 256 + threadIdx.x;
    if (m >= M_NODES) return;
    float v[N2];
    float mx = -3.4e38f;
    float* row = out + (size_t)m * N2;
#pragma unroll
    for (int c = 0; c < N2; ++c) {
        v[c] = row[c] + b2[c];
        mx = fmaxf(mx, v[c]);
    }
    float s = 0.0f;
#pragma unroll
    for (int c = 0; c < N2; ++c) s += __expf(v[c] - mx);
    float l = __logf(s);
#pragma unroll
    for (int c = 0; c < N2; ++c) row[c] = v[c] - mx - l;
}

// ---------------------------------------------------------------------------
extern "C" void kernel_launch(void* const* d_in, const int* in_sizes, int n_in,
                              void* d_out, int out_size, void* d_ws, size_t ws_size,
                              hipStream_t stream) {
    const float* x   = (const float*)d_in[0];
    const float* W1  = (const float*)d_in[1];
    const float* b1  = (const float*)d_in[2];
    const float* W2  = (const float*)d_in[3];
    const float* b2  = (const float*)d_in[4];
    const float* ew  = (const float*)d_in[5];
    const int*   src = (const int*)d_in[6];
    const int*   dst = (const int*)d_in[7];
    float* outf = (float*)d_out;

    // workspace layout (~272 MB)
    char* ws = (char*)d_ws;
    unsigned short* w1t = (unsigned short*)(ws);                         // 262144 B
    unsigned short* w2t = (unsigned short*)(ws + 262144);                // 24576 B
    float*        xw1  = (float*)(ws + 286720);                         // 102.4 MB
    float*        agg1 = (float*)(ws + 286720 + 102400000ull);          // 102.4 MB
    unsigned int* h1   = (unsigned int*)(ws + 286720 + 204800000ull);   // 51.2 MB
    float*        xw2  = (float*)(ws + 286720 + 256000000ull);          // 16 MB

    // zero the scatter targets (d_out doubles as AGG2 accumulator)
    hipMemsetAsync(agg1, 0, (size_t)M_NODES * N1 * sizeof(float), stream);
    hipMemsetAsync(outf, 0, (size_t)M_NODES * N2 * sizeof(float), stream);

    conv_w1_kernel<<<512, 256, 0, stream>>>(W1, w1t);
    conv_w2_kernel<<<48, 256, 0, stream>>>(W2, w2t);

    gemm1_kernel<<<1563, 256, 0, stream>>>(x, (const unsigned int*)w1t, xw1);
    spmm256_kernel<<<800000, 256, 0, stream>>>(src, dst, ew, xw1, agg1);
    bias_relu_pack_kernel<<<50000, 256, 0, stream>>>(agg1, b1, h1);
    gemm2_kernel<<<391, 256, 0, stream>>>(h1, (const unsigned int*)w2t, xw2);
    spmm40_kernel<<<125000, 256, 0, stream>>>(src, dst, ew, xw2, outf);
    logsoftmax_kernel<<<391, 256, 0, stream>>>(outf, b2);
}
